// Decoder_47064251629825
// MI455X (gfx1250) — matrix operand order repaired and checked by reading, not compiled, not run on hardware
//
#include <hip/hip_runtime.h>

typedef _Float16 v16h __attribute__((ext_vector_type(16)));
typedef _Float16 v8h  __attribute__((ext_vector_type(8)));
typedef float    v8f  __attribute__((ext_vector_type(8)));

namespace {
constexpr int kH     = 32;
constexpr int kB     = 2048;
constexpr int kT     = 1024;
constexpr int kTiles = kB / 16;   // 128 workgroups, 2 waves each

__device__ __forceinline__ float fast_rcp(float x) {
  return __builtin_amdgcn_rcpf(x);           // v_rcp_f32, no div expansion
}
__device__ __forceinline__ float fast_sigmoid(float x) {
  return fast_rcp(1.0f + __expf(-x));        // v_exp + v_add + v_rcp
}
__device__ __forceinline__ float fast_tanh(float x) {
  return fmaf(2.0f, fast_rcp(1.0f + __expf(-2.0f * x)), -1.0f);
}
} // namespace

// Two waves per 16-row batch tile.  Wave w owns gate columns j = 16w..16w+15
// (N-tiles p = 2q + w, q = gate type i,f,g,o).  Per lane (hi = lane/16):
//   gate g = 32q + jb + r,  jb = 16w + 8*hi,  r = 0..7
// New h halves are exchanged through LDS (1 ds_store_b128 per lane, barrier,
// one 32B fragment load) -- this doubles wave parallelism per step and
// replaces the shuffle-based fragment transpose.
__global__ __launch_bounds__(64)
void lstm_decoder_kernel(const float* __restrict__ seq,    // (B,T,1)
                         const float* __restrict__ z,      // (B,H)
                         const float* __restrict__ w_ih0,  // (128,1)
                         const float* __restrict__ w_hh0,  // (128,32)
                         const float* __restrict__ b_ih0,
                         const float* __restrict__ b_hh0,
                         const float* __restrict__ w_ih1,  // (128,32)
                         const float* __restrict__ w_hh1,  // (128,32)
                         const float* __restrict__ b_ih1,
                         const float* __restrict__ b_hh1,
                         const float* __restrict__ w_out,  // (1,32)
                         const float* __restrict__ b_out,  // (1,)
                         float* __restrict__ rec,          // (B,T) = d_out+1
                         float* __restrict__ lossWs)       // (kTiles,)
{
  __shared__ _Float16 hbuf0[16][32];   // new h, layer 0 (row-major f16)
  __shared__ _Float16 hbuf1[16][32];   // new h, layer 1
  __shared__ float    predbuf[2][16];  // per-wave partial output projections

  const int lane = threadIdx.x & 31;
  const int w    = threadIdx.x >> 5;   // wave id within tile: 0 or 1
  const int m    = lane & 15;
  const int hi   = lane >> 4;
  const int tile = blockIdx.x;
  const int b    = tile * 16 + m;      // batch row (B/C/D column N = m)
  const int jb   = 16 * w + 8 * hi;    // this lane's gate-column base

  // ---- loop-invariant A-fragments (weight rows, f16) -------------------
  // A 16x32 f16 fragment: lane L -> row M = L%16 of the tile;
// element e -> K = e + 16*(L/16).  Tile q -> rows 16*(2q+w)..
  v16h a_hh0[4], a_ih1[4], a_hh1[4];
#pragma unroll
  for (int q = 0; q < 4; ++q) {
    const int n = 16 * (2 * q + w) + m;
    const float* r0 = w_hh0 + n * kH;
    const float* r1 = w_ih1 + n * kH;
    const float* r2 = w_hh1 + n * kH;
#pragma unroll
    for (int e = 0; e < 8; ++e) {
      a_hh0[q][e]     = (_Float16)r0[16 * hi + e];
      a_hh0[q][e + 8] = (_Float16)r0[16 * hi + 8 + e];
      a_ih1[q][e]     = (_Float16)r1[16 * hi + e];
      a_ih1[q][e + 8] = (_Float16)r1[16 * hi + 8 + e];
      a_hh1[q][e]     = (_Float16)r2[16 * hi + e];
      a_hh1[q][e + 8] = (_Float16)r2[16 * hi + 8 + e];
    }
  }

  // Bias / w_ih0 fragments: value index g = 32q + jb + r
  v8f b0v[4], b1v[4], wi0v[4];
#pragma unroll
  for (int q = 0; q < 4; ++q) {
#pragma unroll
    for (int r = 0; r < 8; ++r) {
      const int g = 32 * q + jb + r;
      b0v[q][r]  = b_ih0[g] + b_hh0[g];
      b1v[q][r]  = b_ih1[g] + b_hh1[g];
      wi0v[q][r] = w_ih0[g];
    }
  }

  // Output projection weights for this lane's j-range
  float wo[8];
#pragma unroll
  for (int r = 0; r < 8; ++r) wo[r] = w_out[jb + r];
  const float bo = b_out[0];

  // ---- initial state: h0 = c0 = h1 = c1 = z ----------------------------
  float c0v[8], c1v[8];
  const float* zr = z + b * kH;
#pragma unroll
  for (int r = 0; r < 8; ++r) {
    c0v[r] = zr[jb + r];
    c1v[r] = c0v[r];
  }
  v16h h0f, h1f;   // B-fragment: element e -> K = e + 16*hi
#pragma unroll
  for (int e = 0; e < 16; ++e) h0f[e] = (_Float16)zr[e + 16 * hi];
  h1f = h0f;

  float pred    = 0.0f;   // replicated in all 64 threads
  float lossAcc = 0.0f;

  // ---- sequential recurrence ------------------------------------------
#pragma unroll 1
  for (int t = 0; t < kT; ++t) {
    v8f acc[4];

    // layer 0: acc = b0 + w_ih0 * pred (rank-1 x-path), then += Whh0 x h0
#pragma unroll
    for (int q = 0; q < 4; ++q)
#pragma unroll
      for (int r = 0; r < 8; ++r)
        acc[q][r] = fmaf(wi0v[q][r], pred, b0v[q][r]);
#pragma unroll
    for (int q = 0; q < 4; ++q)
      acc[q] = __builtin_amdgcn_wmma_f32_16x16x32_f16(
          false, a_hh0[q], false, h0f, (short)0, acc[q], false, false);

    // layer-0 cell update for this lane's 8 gate columns
    float hv[8];
#pragma unroll
    for (int r = 0; r < 8; ++r) {
      const float ig = fast_sigmoid(acc[0][r]);
      const float fg = fast_sigmoid(acc[1][r]);
      const float gg = fast_tanh(acc[2][r]);
      const float og = fast_sigmoid(acc[3][r]);
      const float cn = fmaf(fg, c0v[r], ig * gg);
      c0v[r] = cn;
      hv[r]  = og * fast_tanh(cn);
    }
    {
      v8h hp;
#pragma unroll
      for (int r = 0; r < 8; ++r) hp[r] = (_Float16)hv[r];
      *(v8h*)&hbuf0[m][jb] = hp;            // one ds_store_b128
    }
    __syncthreads();                        // publish h0 halves
    h0f = *(const v16h*)&hbuf0[m][16 * hi]; // full 32B B-fragment

    // layer 1: acc = b1 + Wih1 x h0new + Whh1 x h1
#pragma unroll
    for (int q = 0; q < 4; ++q) acc[q] = b1v[q];
#pragma unroll
    for (int q = 0; q < 4; ++q)
      acc[q] = __builtin_amdgcn_wmma_f32_16x16x32_f16(
          false, a_ih1[q], false, h0f, (short)0, acc[q], false, false);
#pragma unroll
    for (int q = 0; q < 4; ++q)
      acc[q] = __builtin_amdgcn_wmma_f32_16x16x32_f16(
          false, a_hh1[q], false, h1f, (short)0, acc[q], false, false);

#pragma unroll
    for (int r = 0; r < 8; ++r) {
      const float ig = fast_sigmoid(acc[0][r]);
      const float fg = fast_sigmoid(acc[1][r]);
      const float gg = fast_tanh(acc[2][r]);
      const float og = fast_sigmoid(acc[3][r]);
      const float cn = fmaf(fg, c1v[r], ig * gg);
      c1v[r] = cn;
      hv[r]  = og * fast_tanh(cn);
    }
    {
      v8h hp;
#pragma unroll
      for (int r = 0; r < 8; ++r) hp[r] = (_Float16)hv[r];
      *(v8h*)&hbuf1[m][jb] = hp;
    }

    // partial output projection over this lane's j-range (f32)
    float sp = 0.0f;
#pragma unroll
    for (int r = 0; r < 8; ++r) sp = fmaf(hv[r], wo[r], sp);
    sp += __shfl_xor(sp, 16, 32);           // sum over hi halves
    if (hi == 0) predbuf[w][m] = sp;
    __syncthreads();                        // publish h1 halves + partials

    h1f  = *(const v16h*)&hbuf1[m][16 * hi];
    pred = predbuf[0][m] + predbuf[1][m] + bo;

    // emit recovered + accumulate loss (one owner per batch row)
    if (w == 0 && hi == 0) {
      rec[b * kT + t] = pred;
      const float x = seq[b * kT + t];
      const float d = x - pred;
      lossAcc = fmaf(d, d, lossAcc);
    }
  }

  // wave-0 reduction of squared-error partials (lanes 16..31 hold 0)
  if (w == 0) {
#pragma unroll
    for (int off = 16; off >= 1; off >>= 1)
      lossAcc += __shfl_xor(lossAcc, off, 32);
    if (lane == 0) lossWs[tile] = lossAcc;
  }
}

// Deterministic final reduction of the 128 per-tile partials.
__global__ __launch_bounds__(32)
void finalize_loss(const float* __restrict__ ws, float* __restrict__ out) {
  if (threadIdx.x == 0 && blockIdx.x == 0) {
    float s = 0.0f;
    for (int i = 0; i < kTiles; ++i) s += ws[i];
    out[0] = s / (float)(kB * kT);
  }
}

extern "C" void kernel_launch(void* const* d_in, const int* in_sizes, int n_in,
                              void* d_out, int out_size, void* d_ws, size_t ws_size,
                              hipStream_t stream) {
  const float* seq   = (const float*)d_in[0];
  const float* z     = (const float*)d_in[1];
  // d_in[2] = lengths (all == T, unused — matches reference semantics)
  const float* w_ih0 = (const float*)d_in[3];
  const float* w_hh0 = (const float*)d_in[4];
  const float* b_ih0 = (const float*)d_in[5];
  const float* b_hh0 = (const float*)d_in[6];
  const float* w_ih1 = (const float*)d_in[7];
  const float* w_hh1 = (const float*)d_in[8];
  const float* b_ih1 = (const float*)d_in[9];
  const float* b_hh1 = (const float*)d_in[10];
  const float* w_out = (const float*)d_in[11];
  const float* b_out = (const float*)d_in[12];

  float* out    = (float*)d_out;   // [loss, recovered(B*T)]
  float* lossWs = (float*)d_ws;    // kTiles partial sums

  lstm_decoder_kernel<<<kTiles, 64, 0, stream>>>(
      seq, z, w_ih0, w_hh0, b_ih0, b_hh0,
      w_ih1, w_hh1, b_ih1, b_hh1, w_out, b_out,
      out + 1, lossWs);
  finalize_loss<<<1, 32, 0, stream>>>(lossWs, out);
}
